// GAT_19447611916926
// MI455X (gfx1250) — matrix operand order repaired and checked by reading, not compiled, not run on hardware
//
#include <hip/hip_runtime.h>

// ---------------------------------------------------------------------------
// Problem constants (from reference)
// ---------------------------------------------------------------------------
#define NN  50000      // nodes
#define NE  640000     // edges (before self loops)
#define ET  690000     // edges + self loops
#define EPS 1e-16f
#define NEG_SLOPE 0.2f

typedef __attribute__((ext_vector_type(16))) _Float16 v16h;
typedef __attribute__((ext_vector_type(8)))  _Float16 v8h;
typedef __attribute__((ext_vector_type(8)))  float    v8f;

union VPack { v16h v; v8h h[2]; };

// ---------------------------------------------------------------------------
// Helpers
// ---------------------------------------------------------------------------
__device__ __forceinline__ unsigned flipf(float f) {
    unsigned u = __float_as_uint(f);
    return (u & 0x80000000u) ? ~u : (u | 0x80000000u);
}
__device__ __forceinline__ float unflipf(unsigned u) {
    return (u & 0x80000000u) ? __uint_as_float(u & 0x7fffffffu)
                             : __uint_as_float(~u);
}
__device__ __forceinline__ void load_edge(const long long* __restrict__ ei,
                                          int e, int& s, int& d) {
    if (e < NE) { s = (int)ei[e]; d = (int)ei[NE + e]; }
    else        { s = e - NE; d = e - NE; }          // appended self loops
}
__device__ __forceinline__ float lrelu(float v) {
    return v > 0.f ? v : NEG_SLOPE * v;
}

// ---------------------------------------------------------------------------
// Prep kernels
// ---------------------------------------------------------------------------
__global__ void k_cvt_f16(const float* __restrict__ in, _Float16* __restrict__ out, int n) {
    int t = blockIdx.x * blockDim.x + threadIdx.x;
    if (t < n) out[t] = (_Float16)in[t];
}

// in: row-major f32 [K,N]  ->  out: col-major f16 (out[n*K+k])
__global__ void k_transpose_f16(const float* __restrict__ in, _Float16* __restrict__ out,
                                int K, int N) {
    int t = blockIdx.x * blockDim.x + threadIdx.x;
    if (t >= K * N) return;
    int k = t / N, n = t - k * N;
    out[(size_t)n * K + k] = (_Float16)in[t];
}

// ---------------------------------------------------------------------------
// WMMA GEMM:  C[M,NOUT] (f32) = A[M,K] (f16, row major) * BT (f16, col major)
// One wave -> 16 rows x 64 cols strip (4 wmma tiles), K stepped by 32.
// ---------------------------------------------------------------------------
template<int K, int NOUT>
__global__ __launch_bounds__(256)
void k_wmma_gemm(const _Float16* __restrict__ A, const _Float16* __restrict__ BT,
                 float* __restrict__ C, int M)
{
    constexpr int NT = NOUT / 64;                    // 64-wide strips per row group
    const int wid  = (blockIdx.x * blockDim.x + threadIdx.x) >> 5;
    const int lane = threadIdx.x & 31;
    const int tm = wid / NT;
    const int tn = wid % NT;
    if (tm * 16 >= M) return;

    const int half16 = lane >> 4;                    // 0: lanes 0-15, 1: lanes 16-31
    const int l15    = lane & 15;

    const _Float16* arow = A + (size_t)(tm * 16 + l15) * K;

    v8f zero = {0.f,0.f,0.f,0.f,0.f,0.f,0.f,0.f};
    v8f acc[4] = {zero, zero, zero, zero};

    for (int k0 = 0; k0 < K; k0 += 32) {
        // A fragment (ISA 7.12.2): lane holds row M=l15,
        // halves j<8 -> K = k0 + 8*half16 + j ; j>=8 -> K = k0 + 16 + 8*half16 + (j-8)
        VPack ap;
        ap.h[0] = *(const v8h*)(arow + k0 + half16 * 8);
        ap.h[1] = *(const v8h*)(arow + k0 + 16 + half16 * 8);

        #pragma unroll
        for (int t = 0; t < 4; ++t) {
            const int col = tn * 64 + t * 16 + l15;
            // B fragment: lane holds col N=l15, same K mapping as A fragment
            const _Float16* bcol = BT + (size_t)col * K + k0 + half16 * 8;
            VPack bp;
            bp.h[0] = *(const v8h*)(bcol);
            bp.h[1] = *(const v8h*)(bcol + 16);
            acc[t] = __builtin_amdgcn_wmma_f32_16x16x32_f16(
                         false, ap.v, false, bp.v, (short)0, acc[t], false, false);
        }
    }

    // C/D layout: lane -> col N=l15, VGPR r -> row M = 8*half16 + r
    const int mbase = tm * 16 + half16 * 8;
    #pragma unroll
    for (int t = 0; t < 4; ++t) {
        const int col = tn * 64 + t * 16 + l15;
        #pragma unroll
        for (int r = 0; r < 8; ++r)
            C[(size_t)(mbase + r) * NOUT + col] = acc[t][r];
    }
}

// ---------------------------------------------------------------------------
// Per-node attention logits:  al_{src,dst}[n,h] = sum_c h[n,h,c] * a_{src,dst}[h,c]
// ---------------------------------------------------------------------------
template<int H, int OC>
__global__ void k_logits(const float* __restrict__ hfeat,
                         const float* __restrict__ asrc, const float* __restrict__ adst,
                         float* __restrict__ als, float* __restrict__ ald)
{
    int t = blockIdx.x * blockDim.x + threadIdx.x;
    if (t >= NN * H) return;
    int n = t / H, hh = t - n * H;
    const float* row = hfeat + (size_t)n * (H * OC) + hh * OC;
    const float* as  = asrc + hh * OC;
    const float* ad  = adst + hh * OC;
    float s = 0.f, d = 0.f;
    #pragma unroll 8
    for (int c = 0; c < OC; ++c) { float v = row[c]; s += v * as[c]; d += v * ad[c]; }
    als[t] = s; ald[t] = d;
}

// ---------------------------------------------------------------------------
// Edge pass 1: segment max via ordered-uint atomicMax
// ---------------------------------------------------------------------------
template<int H>
__global__ void k_edge_max(const float* __restrict__ als, const float* __restrict__ ald,
                           const long long* __restrict__ ei, unsigned* __restrict__ m)
{
    int t = blockIdx.x * blockDim.x + threadIdx.x;
    if (t >= ET * H) return;
    int e = t / H, h = t - e * H;
    int s, d; load_edge(ei, e, s, d);
    float v = lrelu(als[s * H + h] + ald[d * H + h]);
    atomicMax(m + d * H + h, flipf(v));
}

// ---------------------------------------------------------------------------
// Edge pass 2: ex = exp(e - max[dst]); store ex, atomic sum into s[dst]
// ---------------------------------------------------------------------------
template<int H>
__global__ void k_edge_exp(const float* __restrict__ als, const float* __restrict__ ald,
                           const long long* __restrict__ ei, const unsigned* __restrict__ m,
                           float* __restrict__ ssum, float* __restrict__ ex)
{
    int t = blockIdx.x * blockDim.x + threadIdx.x;
    if (t >= ET * H) return;
    int e = t / H, h = t - e * H;
    int s, d; load_edge(ei, e, s, d);
    float v  = lrelu(als[s * H + h] + ald[d * H + h]);
    float mx = unflipf(m[d * H + h]);
    float ev = expf(v - mx);
    ex[t] = ev;
    atomicAdd(ssum + d * H + h, ev);
}

// ---------------------------------------------------------------------------
// Edge pass 3: agg[dst, c] += feat[src, c] * ex[e,h] / (s[dst,h] + EPS)
// One thread per (edge, channel). C = H << CSHIFT.
// ---------------------------------------------------------------------------
template<int C, int CSHIFT, int H>
__global__ void k_edge_agg(const float* __restrict__ feat, const long long* __restrict__ ei,
                           const float* __restrict__ ex, const float* __restrict__ ssum,
                           float* __restrict__ agg)
{
    long long t = (long long)blockIdx.x * blockDim.x + threadIdx.x;
    if (t >= (long long)ET * C) return;
    int e = (int)(t / C);
    int c = (int)(t - (long long)e * C);
    int h = c >> CSHIFT;
    int sn, dn; load_edge(ei, e, sn, dn);
    float alpha = ex[(size_t)e * H + h] / (ssum[dn * H + h] + EPS);
    atomicAdd(agg + (size_t)dn * C + c, feat[(size_t)sn * C + c] * alpha);
}

// ---------------------------------------------------------------------------
// Layer-1 epilogue: +bias, ELU, emit f16 for next GEMM
// ---------------------------------------------------------------------------
__global__ void k_finalize1(const float* __restrict__ agg, const float* __restrict__ b,
                            _Float16* __restrict__ outh)
{
    int t = blockIdx.x * blockDim.x + threadIdx.x;
    if (t >= NN * 256) return;
    float v = agg[t] + b[t & 255];
    v = v > 0.f ? v : expm1f(v);               // ELU (alpha = 1)
    outh[t] = (_Float16)v;
}

// Layer-2 epilogue: +bias (heads=1 mean is identity), f32 output
__global__ void k_final(const float* __restrict__ agg, const float* __restrict__ b,
                        float* __restrict__ out)
{
    int t = blockIdx.x * blockDim.x + threadIdx.x;
    if (t >= NN * 64) return;
    out[t] = agg[t] + b[t & 63];
}

// ---------------------------------------------------------------------------
// Workspace layout (bytes, 256-aligned, lifetime-aliased; ~144 MB total)
// ---------------------------------------------------------------------------
#define OFF_XH    0ull           // 12,800,000  f16 x           (dead after GEMM1)
#define OFF_H1    12800000ull    // 51,200,000  f32 h1          (dead after pass3)
#define OFF_H1H   0ull           // 25,600,000  f16 elu(h1+b1)  (overlays dead xh/h1)
#define OFF_H2    25600000ull    // 12,800,000  f32 h2
#define OFF_AL2S  38400000ull    //    200,192
#define OFF_AL2D  38600192ull    //    200,192
#define OFF_M2    38800384ull    //    200,192
#define OFF_S2    39000576ull    //    200,192
#define OFF_EX2   39200768ull    //  2,760,192
#define OFF_AGG2  41960960ull    // 12,800,000  (ends 54,760,960 < 64,000,000)
#define OFF_W1T   64000000ull    //     65,536
#define OFF_W2T   64065536ull    //     32,768
#define OFF_AL1S  64098304ull    //  1,600,000
#define OFF_AL1D  65698304ull    //  1,600,000
#define OFF_M1    67298304ull    //  1,600,000
#define OFF_S1    68898304ull    //  1,600,000  (m1+s1 zeroed with one memset)
#define OFF_EX1   70498304ull    // 22,080,000
#define OFF_AGG1  92578304ull    // 51,200,000  (ends 143,778,304)

extern "C" void kernel_launch(void* const* d_in, const int* in_sizes, int n_in,
                              void* d_out, int out_size, void* d_ws, size_t ws_size,
                              hipStream_t stream)
{
    (void)in_sizes; (void)n_in; (void)out_size; (void)ws_size;

    const float*     x    = (const float*)d_in[0];
    const long long* ei   = (const long long*)d_in[1];   // int64 [2, NE]
    const float*     W1   = (const float*)d_in[2];       // [128,256]
    const float*     a1s  = (const float*)d_in[3];       // [8,32]
    const float*     a1d  = (const float*)d_in[4];
    const float*     b1   = (const float*)d_in[5];       // [256]
    const float*     W2   = (const float*)d_in[6];       // [256,64]
    const float*     a2s  = (const float*)d_in[7];       // [1,64]
    const float*     a2d  = (const float*)d_in[8];
    const float*     b2   = (const float*)d_in[9];       // [64]
    float*           out  = (float*)d_out;               // [NN,64]

    char* ws = (char*)d_ws;
    _Float16* xh   = (_Float16*)(ws + OFF_XH);
    float*    h1   = (float*)   (ws + OFF_H1);
    _Float16* h1h  = (_Float16*)(ws + OFF_H1H);
    float*    h2   = (float*)   (ws + OFF_H2);
    float*    al2s = (float*)   (ws + OFF_AL2S);
    float*    al2d = (float*)   (ws + OFF_AL2D);
    unsigned* m2   = (unsigned*)(ws + OFF_M2);
    float*    s2   = (float*)   (ws + OFF_S2);
    float*    ex2  = (float*)   (ws + OFF_EX2);
    float*    agg2 = (float*)   (ws + OFF_AGG2);
    _Float16* w1t  = (_Float16*)(ws + OFF_W1T);
    _Float16* w2t  = (_Float16*)(ws + OFF_W2T);
    float*    al1s = (float*)   (ws + OFF_AL1S);
    float*    al1d = (float*)   (ws + OFF_AL1D);
    unsigned* m1   = (unsigned*)(ws + OFF_M1);
    float*    s1   = (float*)   (ws + OFF_S1);
    float*    ex1  = (float*)   (ws + OFF_EX1);
    float*    agg1 = (float*)   (ws + OFF_AGG1);

    const int B = 256;
    #define GRID(n) dim3((unsigned)(((long long)(n) + B - 1) / B))

    // ---- prep: f16 conversions / weight transposes ----
    k_cvt_f16      <<<GRID(NN * 128), B, 0, stream>>>(x, xh, NN * 128);
    k_transpose_f16<<<GRID(128 * 256), B, 0, stream>>>(W1, w1t, 128, 256);
    k_transpose_f16<<<GRID(256 * 64),  B, 0, stream>>>(W2, w2t, 256, 64);

    // ---- layer 1 ----
    hipMemsetAsync(ws + OFF_M1,   0, 3200000,  stream);   // m1 + s1
    hipMemsetAsync(ws + OFF_AGG1, 0, 51200000, stream);   // agg1

    // GEMM1: waves = (NN/16) * (256/64) = 12500, 8 waves/block
    k_wmma_gemm<128, 256><<<dim3(1563), B, 0, stream>>>(xh, w1t, h1, NN);
    k_logits<8, 32><<<GRID(NN * 8), B, 0, stream>>>(h1, a1s, a1d, al1s, al1d);

    k_edge_max<8><<<GRID(ET * 8), B, 0, stream>>>(al1s, al1d, ei, m1);
    k_edge_exp<8><<<GRID(ET * 8), B, 0, stream>>>(al1s, al1d, ei, m1, s1, ex1);
    k_edge_agg<256, 5, 8><<<GRID((long long)ET * 256), B, 0, stream>>>(h1, ei, ex1, s1, agg1);

    k_finalize1<<<GRID(NN * 256), B, 0, stream>>>(agg1, b1, h1h);

    // ---- layer 2 ----
    hipMemsetAsync(ws + OFF_M2,   0, 400384,   stream);   // m2 + s2
    hipMemsetAsync(ws + OFF_AGG2, 0, 12800000, stream);   // agg2

    // GEMM2: waves = NN/16 = 3125, 8 waves/block
    k_wmma_gemm<256, 64><<<dim3(391), B, 0, stream>>>(h1h, w2t, h2, NN);
    k_logits<1, 64><<<GRID(NN), B, 0, stream>>>(h2, a2s, a2d, al2s, al2d);

    k_edge_max<1><<<GRID(ET), B, 0, stream>>>(al2s, al2d, ei, m2);
    k_edge_exp<1><<<GRID(ET), B, 0, stream>>>(al2s, al2d, ei, m2, s2, ex2);
    k_edge_agg<64, 6, 1><<<GRID((long long)ET * 64), B, 0, stream>>>(h2, ei, ex2, s2, agg2);

    k_final<<<GRID(NN * 64), B, 0, stream>>>(agg2, b2, out);

    #undef GRID
}
